// LocalAttention_4758823764713
// MI455X (gfx1250) — matrix operand order repaired and checked by reading, not compiled, not run on hardware
//
#include <hip/hip_runtime.h>
#include <hip/hip_bf16.h>
#include <stddef.h>

// ---------------------------------------------------------------------------
// Local attention: N=8, C_IN=128, C_OUT=64, H=W=128, KH=KW=9, DILATE=2
//   q,k,v = 1x1 convs  : TDM(B panel -> LDS) + v_wmma_f32_16x16x32_bf16
//   k,v stored into zero-padded 144x144 planes (reference zero-pad semantics)
//   logits/softmax/weighting : mask-free VALU streaming from L2-resident k,v
// ---------------------------------------------------------------------------

#define NB     8
#define CIN    128
#define COUT   64
#define HH_    128
#define WW_    128
#define HWSZ   (HH_*WW_)          // 16384
#define NPIX   (NB*HWSZ)          // 131072
#define KWIN   9
#define DIL    2
#define NOFF   (KWIN*KWIN)        // 81
#define PAD    8                  // dilate * (KWIN/2) = 8
#define PW     (WW_ + 2*PAD)      // 144
#define PLANE  (PW*PW)            // 20736  (padded per-(n,c) plane)

typedef __attribute__((ext_vector_type(8)))  __bf16 v8bf;
typedef __attribute__((ext_vector_type(16))) __bf16 v16bf;
typedef __attribute__((ext_vector_type(8)))  float  v8f;
typedef __attribute__((ext_vector_type(4)))  unsigned int u32x4;
typedef __attribute__((ext_vector_type(8)))  int i32x8;
typedef __attribute__((ext_vector_type(4)))  int i32x4;

union V16 { v16bf v; v8bf h[2]; };

__device__ __forceinline__ unsigned short f2bf(float f) {
    union { float f; unsigned u; } x; x.f = f;
    unsigned u = x.u + 0x7fffu + ((x.u >> 16) & 1u);   // round to nearest even
    return (unsigned short)(u >> 16);
}
__device__ __forceinline__ float bf2f(unsigned short s) {
    union { unsigned u; float f; } x; x.u = ((unsigned)s) << 16;
    return x.f;
}

// ---------------------------------------------------------------------------
// Kernel 1: x [N,CIN,HW] f32  ->  xbf [N*HW, CIN] bf16 (pixel-major), via LDS
// ---------------------------------------------------------------------------
__global__ __launch_bounds__(256) void pack_x(const float* __restrict__ x,
                                              unsigned short* __restrict__ xbf) {
    __shared__ float tile[32][33];
    const int n   = blockIdx.z;
    const int hw0 = blockIdx.x * 32;
    const int c0  = blockIdx.y * 32;
    const int tx  = threadIdx.x;     // 0..31
    const int ty  = threadIdx.y;     // 0..7
#pragma unroll
    for (int j = 0; j < 32; j += 8)
        tile[ty + j][tx] = x[(((size_t)(n * CIN + c0 + ty + j)) << 14) + hw0 + tx];
    __syncthreads();
#pragma unroll
    for (int j = 0; j < 32; j += 8) {
        const int hwi = hw0 + ty + j;
        xbf[(((size_t)(n << 14) + hwi) << 7) + c0 + tx] = f2bf(tile[tx][ty + j]);
    }
}

// ---------------------------------------------------------------------------
// Kernel 2: W1,W2,W3 [64,128] f32 -> wbf [3,64,128] bf16
// ---------------------------------------------------------------------------
__global__ __launch_bounds__(256) void pack_w(const float* __restrict__ W1,
                                              const float* __restrict__ W2,
                                              const float* __restrict__ W3,
                                              unsigned short* __restrict__ wbf) {
    const int i = blockIdx.x * 256 + threadIdx.x;   // 0 .. 24575
    const int per = COUT * CIN;                     // 8192
    if (i < per)            wbf[i] = f2bf(W1[i]);
    else if (i < 2 * per)   wbf[i] = f2bf(W2[i - per]);
    else                    wbf[i] = f2bf(W3[i - 2 * per]);
}

// ---------------------------------------------------------------------------
// Kernel 3: zero a buffer (for the padded k/v halos), 16B per thread.
// ---------------------------------------------------------------------------
__global__ __launch_bounds__(256) void zero_buf(uint4* __restrict__ p) {
    const size_t i = (size_t)blockIdx.x * 256 + threadIdx.x;
    p[i] = uint4{0u, 0u, 0u, 0u};
}

// ---------------------------------------------------------------------------
// Kernel 4: fused 3-way GEMM.
//   TDM stages the 32-pixel x 128-ch bf16 B-panel (8 KB, contiguous) to LDS;
//   8 waves (2 pixel-tiles x 4 co-tiles) run v_wmma_f32_16x16x32_bf16, K=128.
//   q written [N,64,HW] bf16; k,v written into zero-padded 144x144 planes.
// ---------------------------------------------------------------------------
__global__ __launch_bounds__(256) void conv_wmma(const unsigned short* __restrict__ xbf,
                                                 const unsigned short* __restrict__ wbf,
                                                 unsigned short* __restrict__ qb,
                                                 unsigned short* __restrict__ kpad,
                                                 unsigned short* __restrict__ vpad) {
    __shared__ unsigned short Bs[32 * CIN];          // 8 KB B panel
    const int wv   = threadIdx.x >> 5;               // wave 0..7
    const int lane = threadIdx.x & 31;
    const int P0   = blockIdx.x * 32;                // block pixel base

    if (wv == 0) {
        // --- Tensor Data Mover: 1D copy of 8192 B (2048 dwords) global->LDS
        const unsigned lds_off = (unsigned)(unsigned long long)(&Bs[0]);
        const unsigned long long ga = (unsigned long long)(xbf + (size_t)P0 * CIN);
        u32x4 g0 = {};
        g0[0] = 1u;                                          // count=1 (user D#)
        g0[1] = lds_off;                                     // lds_addr (bytes)
        g0[2] = (unsigned)(ga & 0xffffffffu);                // global_addr lo
        g0[3] = (unsigned)((ga >> 32) & 0x01ffffffu) | (2u << 30);  // hi | type=2
        i32x8 g1 = {};
        const unsigned dim0 = 2048u;                         // dwords (data_size=4B)
        g1[0] = (int)(2u << 16);                             // data_size=2 -> 4B
        g1[1] = (int)((dim0 & 0xffffu) << 16);               // tensor_dim0[15:0]
        g1[2] = (int)((dim0 >> 16) | (1u << 16));            // td0[31:16] | tensor_dim1=1
        g1[3] = (int)((dim0 & 0xffffu) << 16);               // td1[31:16]=0 | tile_dim0
        g1[4] = 1;                                           // tile_dim1=1, tile_dim2=0
        g1[5] = (int)dim0;                                   // tensor_dim0_stride lo32
        g1[6] = 0;                                           // stride hi | td1_stride lo
        g1[7] = 0;
        i32x4 z4 = {};
#if __clang_major__ >= 23
        i32x8 z8 = {};
        __builtin_amdgcn_tensor_load_to_lds(g0, g1, z4, z4, z8, 0);
#else
        __builtin_amdgcn_tensor_load_to_lds(g0, g1, z4, z4, 0);
#endif
        __builtin_amdgcn_s_wait_tensorcnt(0);
    }
    __syncthreads();

    const int pt2 = wv >> 2;           // 0..1 : pixel sub-tile
    const int ct  = wv & 3;            // 0..3 : co tile
    const int col = lane & 15;
    const int hi  = (lane >> 4) & 1;

    const int plocal = pt2 * 16 + col;               // B column pixel (local)
    const int p      = P0 + plocal;                  // global pixel

    const unsigned short* bpl = &Bs[plocal * CIN];                   // LDS row
    const unsigned short* ap  = wbf + (size_t)(ct * 16 + col) * CIN; // W row

    v8f acc1 = {}; v8f acc2 = {}; v8f acc3 = {};

#pragma unroll
    for (int kc = 0; kc < 4; kc++) {
        const int k0 = kc * 32;
        // B 32x16 bf16: lane<16 -> col=lane, K {k0+0..7, k0+16..23}; lane>=16 -> +8
        V16 B;
        const unsigned short* bq = bpl + k0 + hi * 8;
        B.h[0] = *(const v8bf*)(bq);                  // ds_load_b128
        B.h[1] = *(const v8bf*)(bq + 16);
        // A 16x32 bf16: lane<16 -> K {k0+0..7, k0+16..23}; lane>=16 -> +8
        const unsigned short* a1 = ap + k0 + hi * 8;
        V16 A1, A2, A3;
        A1.h[0] = *(const v8bf*)(a1);                  A1.h[1] = *(const v8bf*)(a1 + 16);
        A2.h[0] = *(const v8bf*)(a1 + 1 * COUT * CIN); A2.h[1] = *(const v8bf*)(a1 + 1 * COUT * CIN + 16);
        A3.h[0] = *(const v8bf*)(a1 + 2 * COUT * CIN); A3.h[1] = *(const v8bf*)(a1 + 2 * COUT * CIN + 16);

        acc1 = __builtin_amdgcn_wmma_f32_16x16x32_bf16(false, A1.v, false, B.v, (short)0, acc1, false, false);
        acc2 = __builtin_amdgcn_wmma_f32_16x16x32_bf16(false, A2.v, false, B.v, (short)0, acc2, false, false);
        acc3 = __builtin_amdgcn_wmma_f32_16x16x32_bf16(false, A3.v, false, B.v, (short)0, acc3, false, false);
    }

    // D layout: VGPR r -> row M = r + hi*8 (co), col N = lane&15 (pixel p)
    const int n  = p >> 14;
    const int hw = p & (HWSZ - 1);
    const int h  = hw >> 7;
    const int w  = hw & (WW_ - 1);
    const size_t qbase = ((size_t)(n * COUT + ct * 16) << 14) + hw;
    const size_t pbase = (size_t)(n * COUT + ct * 16) * PLANE + (size_t)(h + PAD) * PW + (w + PAD);
#pragma unroll
    for (int r = 0; r < 8; r++) {
        const int M = r + hi * 8;
        qb[qbase + ((size_t)M << 14)]      = f2bf(acc1[r]);
        kpad[pbase + (size_t)M * PLANE]    = f2bf(acc2[r]);
        vpad[pbase + (size_t)M * PLANE]    = f2bf(acc3[r]);
    }
}

// ---------------------------------------------------------------------------
// Kernel 5: fused local-similar + softmax + local-weighting, mask-free.
// Zero-padded k/v planes reproduce the reference exactly (OOB logits == 0
// and participate in the softmax; OOB v contributes 0).
// ---------------------------------------------------------------------------
__global__ __launch_bounds__(256) void attn_fused(const unsigned short* __restrict__ qb,
                                                  const unsigned short* __restrict__ kpad,
                                                  const unsigned short* __restrict__ vpad,
                                                  float* __restrict__ out) {
    const int pid = blockIdx.x * 256 + threadIdx.x;   // 0 .. NPIX-1
    const int n   = pid >> 14;
    const int hw  = pid & (HWSZ - 1);
    const int h   = hw >> 7;
    const int w   = hw & (WW_ - 1);

    float logits[NOFF];
#pragma unroll
    for (int t = 0; t < NOFF; t++) logits[t] = 0.0f;

    const unsigned short* qp = qb   + ((size_t)(n * COUT) << 14) + hw;
    const unsigned short* kp = kpad + (size_t)(n * COUT) * PLANE + (size_t)(h + PAD) * PW + (w + PAD);

    for (int c = 0; c < COUT; c++) {
        const unsigned short* kc = kp + (size_t)c * PLANE;
        __builtin_prefetch(kc + PLANE, 0, 0);          // global_prefetch_b8
        const float qc = bf2f(qp[(size_t)c << 14]);
#pragma unroll
        for (int t = 0; t < NOFF; t++) {
            const int Ct = ((t / KWIN - 4) * DIL) * PW + (t % KWIN - 4) * DIL;  // compile-time
            logits[t] = fmaf(qc, bf2f(kc[Ct]), logits[t]);
        }
    }

    // softmax over all 81 (padded entries are exactly 0 and join the softmax)
    float mx = logits[0];
#pragma unroll
    for (int t = 1; t < NOFF; t++) mx = fmaxf(mx, logits[t]);
    float s = 0.0f;
#pragma unroll
    for (int t = 0; t < NOFF; t++) { const float e = __expf(logits[t] - mx); logits[t] = e; s += e; }
    const float inv = 1.0f / s;
#pragma unroll
    for (int t = 0; t < NOFF; t++) logits[t] *= inv;

    const unsigned short* vp = vpad + (size_t)(n * COUT) * PLANE + (size_t)(h + PAD) * PW + (w + PAD);
    float* op = out + ((size_t)(n * COUT) << 14) + hw;
    for (int c = 0; c < COUT; c++) {
        const unsigned short* vc = vp + (size_t)c * PLANE;
        __builtin_prefetch(vc + PLANE, 0, 0);
        float acc = 0.0f;
#pragma unroll
        for (int t = 0; t < NOFF; t++) {
            const int Ct = ((t / KWIN - 4) * DIL) * PW + (t % KWIN - 4) * DIL;
            acc = fmaf(logits[t], bf2f(vc[Ct]), acc);
        }
        op[(size_t)c << 14] = acc;                     // coalesced f32 store
    }
}

// ---------------------------------------------------------------------------
extern "C" void kernel_launch(void* const* d_in, const int* in_sizes, int n_in,
                              void* d_out, int out_size, void* d_ws, size_t ws_size,
                              hipStream_t stream) {
    const float* x  = (const float*)d_in[0];
    const float* W1 = (const float*)d_in[1];
    const float* W2 = (const float*)d_in[2];
    const float* W3 = (const float*)d_in[3];
    float* out = (float*)d_out;

    char* ws = (char*)d_ws;
    size_t off = 0;
    auto carve = [&](size_t bytes) {
        char* p = ws + off;
        off += (bytes + 255) & ~(size_t)255;
        return p;
    };
    const size_t padBytes = (size_t)NB * COUT * PLANE * 2;         // 21,233,664 (16B multiple)
    unsigned short* xbf  = (unsigned short*)carve((size_t)NPIX * CIN * 2);
    unsigned short* wbf  = (unsigned short*)carve((size_t)3 * COUT * CIN * 2);
    unsigned short* qb   = (unsigned short*)carve((size_t)NPIX * COUT * 2);
    unsigned short* kpad = (unsigned short*)carve(padBytes);
    unsigned short* vpad = (unsigned short*)carve(padBytes);
    (void)ws_size; (void)in_sizes; (void)n_in; (void)out_size;

    const int zeroBlocks = (int)(padBytes / 16 / 256);             // 5184, exact
    pack_x<<<dim3(HWSZ / 32, CIN / 32, NB), dim3(32, 8, 1), 0, stream>>>(x, xbf);
    pack_w<<<(3 * COUT * CIN) / 256, 256, 0, stream>>>(W1, W2, W3, wbf);
    zero_buf<<<zeroBlocks, 256, 0, stream>>>((uint4*)kpad);
    zero_buf<<<zeroBlocks, 256, 0, stream>>>((uint4*)vpad);
    conv_wmma<<<NPIX / 32, 256, 0, stream>>>(xbf, wbf, qb, kpad, vpad);
    attn_fused<<<NPIX / 256, 256, 0, stream>>>(qb, kpad, vpad, out);
}
